// VanillaSLSTMFeat_36275293782364
// MI455X (gfx1250) — matrix operand order repaired and checked by reading, not compiled, not run on hardware
//
#include <hip/hip_runtime.h>
#include <hip/hip_bf16.h>
#include <cstdint>
#include <cstddef>

// ---------------------------------------------------------------------------
// Types for CDNA5 WMMA (wave32): v16bf A/B fragments, v8f accumulator.
// ---------------------------------------------------------------------------
typedef __bf16 v16bf __attribute__((ext_vector_type(16)));
typedef float  v8f   __attribute__((ext_vector_type(8)));
typedef unsigned short ushort8 __attribute__((ext_vector_type(8)));

struct U16x16 { ushort8 lo; ushort8 hi; };   // 32 bytes == v16bf

static __device__ __forceinline__ unsigned short f2bf(float f) {
  // round-to-nearest-even f32 -> bf16 (NaN edge cases ignored)
  unsigned int u = __float_as_uint(f);
  u += 0x7FFFu + ((u >> 16) & 1u);
  return (unsigned short)(u >> 16);
}

// Async global->LDS copies (ASYNCcnt-tracked). INST_OFFSET is added to BOTH
// the LDS (VDST) and global (VADDR) addresses, so one base pair covers a
// contiguous run of b128 transfers.
static __device__ __forceinline__ void async_copy_32B(unsigned lds,
                                                      unsigned long long ga) {
  asm volatile("global_load_async_to_lds_b128 %0, %1, off\n\t"
               "global_load_async_to_lds_b128 %0, %1, off offset:16"
               :: "v"(lds), "v"(ga) : "memory");
}
static __device__ __forceinline__ void async_copy_64B(unsigned lds,
                                                      unsigned long long ga) {
  asm volatile("global_load_async_to_lds_b128 %0, %1, off\n\t"
               "global_load_async_to_lds_b128 %0, %1, off offset:16\n\t"
               "global_load_async_to_lds_b128 %0, %1, off offset:32\n\t"
               "global_load_async_to_lds_b128 %0, %1, off offset:48"
               :: "v"(lds), "v"(ga) : "memory");
}
static __device__ __forceinline__ void wait_async0() {
  asm volatile("s_wait_asynccnt 0" ::: "memory");
}

// ---------------------------------------------------------------------------
// GEMM: out[M][N] = A[M][K](bf16) @ Bt[N][K]^T(bf16)  + epilogue
//   mode 0: outF = acc                                   (we_cat)
//   mode 1: outF = acc + bias[n] + addend[m][n]          (gates)
//   mode 2: outB = bf16(relu(acc + bias[n]))             (FFN layer 1)
//   mode 3: outF = (acc + bias[n] + addend[m][n]) * sm   (FFN layer 2 + residual + mask)
// Block: 256 threads = 8 wave32, tile 128(M) x 256(N), BK = 32.
// Waves arranged 2(M) x 4(N); each wave owns 64x64 = 4x4 WMMA 16x16 tiles
// -> 16 v_wmma per K-step against 16 fragment ds_load_b128 (1.0 ratio).
// LDS double-buffered (48 KB), filled with GLOBAL_LOAD_ASYNC_TO_LDS_B128;
// one workgroup barrier pair per K-step.
// ---------------------------------------------------------------------------
#define BM 128
#define BN 256
#define BK 32

__global__ __launch_bounds__(256) void gemm_bf16_wmma(
    const unsigned short* __restrict__ A,    // [M][K] bf16 row-major
    const unsigned short* __restrict__ Bt,   // [N][K] bf16 (B transposed, N-major)
    float*          __restrict__ outF,
    unsigned short* __restrict__ outB,
    const float*    __restrict__ bias,       // [N] f32 or null
    const float*    __restrict__ addend,     // [M][N] f32 or null
    const int*      __restrict__ rowmask,    // [M] int (1 = masked row) or null
    int M, int N, int K, int mode)
{
  __shared__ __align__(16) unsigned short As[2][BM * BK];  // 2 x 8 KB
  __shared__ __align__(16) unsigned short Bs[2][BN * BK];  // 2 x 16 KB

  const int tid  = threadIdx.x;
  const int lane = tid & 31;
  const int wave = tid >> 5;
  const int wm   = wave >> 2;      // 0..1  (64-row half)
  const int wn   = wave & 3;       // 0..3  (64-col strip)
  const int kg   = lane >> 4;      // lane group 0/1 (K-half selector)
  const int lr   = lane & 15;      // row/col within 16

  const int bm = blockIdx.y * BM;
  const int bn = blockIdx.x * BN;

  // tile fill mapping: A = 32B/thread (128 rows x 64B), B = 64B/thread (256 rows)
  const int arow = tid >> 1;            // 0..127
  const int acol = (tid & 1) * 16;      // 0 or 16 (halves)
  const int brow = tid;                 // 0..255  (n within tile)

  v8f acc[4][4];
  #pragma unroll
  for (int i = 0; i < 4; ++i)
    #pragma unroll
    for (int j = 0; j < 4; ++j)
      #pragma unroll
      for (int r = 0; r < 8; ++r) acc[i][j][r] = 0.f;

  auto issue_tile = [&](int k0, int buf) {
    async_copy_32B((unsigned)(uintptr_t)&As[buf][arow * BK + acol],
                   (unsigned long long)&A[(size_t)(bm + arow) * K + k0 + acol]);
    async_copy_64B((unsigned)(uintptr_t)&Bs[buf][brow * BK],
                   (unsigned long long)&Bt[(size_t)(bn + brow) * K + k0]);
  };

  const int steps = K / BK;
  issue_tile(0, 0);

  for (int s = 0; s < steps; ++s) {
    wait_async0();       // this wave's fills for buffer s&1 are in LDS
    __syncthreads();     // everyone's fills done; everyone done reading s^1
    if (s + 1 < steps) issue_tile((s + 1) * BK, (s + 1) & 1);
    const int buf = s & 1;

    // Fragments per ISA 7.12.2 wave32 layouts.
    // A 16x32 bf16: lane -> M=lane%16 ; half j: k = (lane/16)*8 + j  (j<8)
    //                                        k = 16 + (lane/16)*8 + (j-8)
    v16bf afrag[4], bfrag[4];
    #pragma unroll
    for (int im = 0; im < 4; ++im) {
      const int r = wm * 64 + im * 16 + lr;
      U16x16 t;
      t.lo = *(const ushort8*)&As[buf][r * BK + kg * 8];
      t.hi = *(const ushort8*)&As[buf][r * BK + 16 + kg * 8];
      afrag[im] = __builtin_bit_cast(v16bf, t);
    }
    // B 32x16 bf16: lane -> N=lane%16 ; half j: k = (lane/16)*8 + j (j<8), k = 16 + (lane/16)*8 + (j-8)
    #pragma unroll
    for (int in = 0; in < 4; ++in) {
      const int n = wn * 64 + in * 16 + lr;
      U16x16 t;
      t.lo = *(const ushort8*)&Bs[buf][n * BK + kg * 8];
      t.hi = *(const ushort8*)&Bs[buf][n * BK + 16 + kg * 8];
      bfrag[in] = __builtin_bit_cast(v16bf, t);
    }
    #pragma unroll
    for (int im = 0; im < 4; ++im)
      #pragma unroll
      for (int in = 0; in < 4; ++in)
        acc[im][in] = __builtin_amdgcn_wmma_f32_16x16x32_bf16(
            false, afrag[im], false, bfrag[in],
            (short)0, acc[im][in], false, false);

    __syncthreads();     // done reading buf before it is refilled at s+2
  }

  // Epilogue: C/D layout -> VGPR r: m = r + (lane/16)*8, n = lane%16
  #pragma unroll
  for (int im = 0; im < 4; ++im) {
    #pragma unroll
    for (int in = 0; in < 4; ++in) {
      const int n = bn + wn * 64 + in * 16 + lr;
      #pragma unroll
      for (int r = 0; r < 8; ++r) {
        const int m = bm + wm * 64 + im * 16 + kg * 8 + r;
        const size_t o = (size_t)m * N + n;
        float v = acc[im][in][r];
        if (mode == 0) {
          outF[o] = v;
        } else if (mode == 1) {
          outF[o] = v + bias[n] + addend[o];
        } else if (mode == 2) {
          float t = v + bias[n];
          outB[o] = f2bf(t > 0.f ? t : 0.f);
        } else {
          float sm = 1.f - (float)rowmask[m];
          outF[o] = (v + bias[n] + addend[o]) * sm;
        }
      }
    }
  }
}

// ---------------------------------------------------------------------------
// Weight prep: dst[n][kOff + k] (bf16, row stride dstStride) = src[k][n] (f32)
// ---------------------------------------------------------------------------
__global__ void transpose_to_bf16(const float* __restrict__ src,
                                  unsigned short* __restrict__ dst,
                                  int K, int N, int dstStride, int kOff)
{
  size_t t = (size_t)blockIdx.x * 256 + threadIdx.x;
  if (t >= (size_t)K * N) return;
  int k = (int)(t / N);
  int n = (int)(t % N);
  dst[(size_t)n * dstStride + kOff + k] = f2bf(src[(size_t)k * N + n]);
}

// ---------------------------------------------------------------------------
// Init: x = word * seq_mask ; h = c = x ; x_bf16 for we_cat GEMM
// ---------------------------------------------------------------------------
__global__ void init_states(const float* __restrict__ x, const int* __restrict__ mask,
                            float* __restrict__ h, float* __restrict__ c,
                            unsigned short* __restrict__ xb, int total)
{
  int i = blockIdx.x * 256 + threadIdx.x;
  if (i >= total) return;
  int m = i >> 9;                       // / 512
  float sm = 1.f - (float)mask[m];
  float v = x[i] * sm;
  h[i] = v; c[i] = v; xb[i] = f2bf(v);
}

// ---------------------------------------------------------------------------
// Shift sums (KERNEL_SIZE=2) + pack A = [h | h_before | h_after | c] as bf16.
// Also emits c_before / c_after in fp32 for the gate kernel.
// grid (2, 8192), block 256. m = b*1024+s so in-batch bounds == s bounds.
// ---------------------------------------------------------------------------
__global__ __launch_bounds__(256) void shift_pack(
    const float* __restrict__ h, const float* __restrict__ c,
    const int* __restrict__ mask,
    unsigned short* __restrict__ Abuf,     // [8192][2048] bf16
    float* __restrict__ cb, float* __restrict__ ca)
{
  const int col = blockIdx.x * 256 + threadIdx.x;   // 0..511
  const int m   = blockIdx.y;                       // 0..8191
  const int s   = m & 1023;
  const float sm = 1.f - (float)mask[m];
  const size_t idx = (size_t)m * 512 + col;

  float hb = 0.f, ha = 0.f, cbv = 0.f, cav = 0.f;
  #pragma unroll
  for (int st = 1; st <= 2; ++st) {
    if (s - st >= 0)  { hb += h[idx - (size_t)st * 512]; cbv += c[idx - (size_t)st * 512]; }
    if (s + st < 1024){ ha += h[idx + (size_t)st * 512]; cav += c[idx + (size_t)st * 512]; }
  }
  hb *= sm; ha *= sm; cbv *= sm; cav *= sm;

  const size_t ar = (size_t)m * 2048;
  Abuf[ar + col]        = f2bf(h[idx]);
  Abuf[ar + 512 + col]  = f2bf(hb);
  Abuf[ar + 1024 + col] = f2bf(ha);
  Abuf[ar + 1536 + col] = f2bf(c[idx]);
  cb[idx] = cbv; ca[idx] = cav;
}

// ---------------------------------------------------------------------------
// Gates (sigmoid->softmax over 5, tanh, output gate), c_new, h_new, LayerNorm.
// One block (256 threads) per row; each thread owns 2 of the 512 columns.
// Writes: c (in-place, masked), lnF (f32), lnB (bf16) for the FFN GEMMs.
// ---------------------------------------------------------------------------
__global__ __launch_bounds__(256) void gate_ln(
    const float* __restrict__ gates,        // [8192][3584]
    float* __restrict__ c,                  // [8192][512] in-out
    const float* __restrict__ cb, const float* __restrict__ ca,
    const int* __restrict__ mask,
    const float* __restrict__ gamma, const float* __restrict__ beta,
    float* __restrict__ lnF, unsigned short* __restrict__ lnB)
{
  __shared__ float rsum[256];
  __shared__ float rsq[256];
  const int m = blockIdx.x;
  const int tid = threadIdx.x;
  const float sm = 1.f - (float)mask[m];

  float hn[2];
  float lsum = 0.f, lsq = 0.f;
  #pragma unroll
  for (int q = 0; q < 2; ++q) {
    const int col = tid + q * 256;
    const size_t g = (size_t)m * 3584 + col;
    float fl = gates[g];
    float fr = gates[g + 512];
    float fc = gates[g + 1024];
    float ft = gates[g + 1536];
    float fi = gates[g + 2048];
    float fw = gates[g + 2560];
    float fo = gates[g + 3072];
    // sigmoid then 5-way softmax (g_feat computed but unused, per reference)
    float s0 = 1.f / (1.f + __expf(-fl));
    float s1 = 1.f / (1.f + __expf(-fr));
    float s2 = 1.f / (1.f + __expf(-fc));
    float s3 = 1.f / (1.f + __expf(-ft));
    float s4 = 1.f / (1.f + __expf(-fi));
    float e0 = __expf(s0), e1 = __expf(s1), e2 = __expf(s2),
          e3 = __expf(s3), e4 = __expf(s4);
    float inv = 1.f / (e0 + e1 + e2 + e3 + e4);
    float gw = tanhf(fw);
    float go = 1.f / (1.f + __expf(-fo));
    const size_t idx = (size_t)m * 512 + col;
    float cn = cb[idx] * (e0 * inv) + ca[idx] * (e1 * inv)
             + c[idx]  * (e2 * inv) + gw * (e4 * inv);
    float h1 = go * tanhf(cn);
    c[idx] = cn * sm;            // masked cell state for next layer
    hn[q] = h1;
    lsum += h1; lsq += h1 * h1;
  }
  rsum[tid] = lsum; rsq[tid] = lsq;
  __syncthreads();
  for (int off = 128; off > 0; off >>= 1) {
    if (tid < off) { rsum[tid] += rsum[tid + off]; rsq[tid] += rsq[tid + off]; }
    __syncthreads();
  }
  const float mu  = rsum[0] * (1.f / 512.f);
  const float var = rsq[0] * (1.f / 512.f) - mu * mu;
  const float rstd = rsqrtf(var + 1e-5f);
  #pragma unroll
  for (int q = 0; q < 2; ++q) {
    const int col = tid + q * 256;
    const size_t idx = (size_t)m * 512 + col;
    float ln = (hn[q] - mu) * rstd * gamma[col] + beta[col];
    lnF[idx] = ln;
    lnB[idx] = f2bf(ln);
  }
}

__global__ void copy_out(const float* __restrict__ src, float* __restrict__ dst, int total)
{
  int i = blockIdx.x * 256 + threadIdx.x;
  if (i < total) dst[i] = src[i];
}

// ---------------------------------------------------------------------------
// Host launch
// ---------------------------------------------------------------------------
extern "C" void kernel_launch(void* const* d_in, const int* in_sizes, int n_in,
                              void* d_out, int out_size, void* d_ws, size_t ws_size,
                              hipStream_t stream)
{
  (void)in_sizes; (void)n_in; (void)out_size; (void)ws_size;

  const float* word  = (const float*)d_in[0];
  const int*   mask  = (const int*)d_in[1];
  // d_in[2] = num_layers (setup fixes it at 4; hardcoded below — device scalar
  // cannot be read host-side under graph capture without a sync)
  const float* Wg    = (const float*)d_in[3];
  const float* bg    = (const float*)d_in[4];
  const float* We    = (const float*)d_in[5];
  const float* Wp    = (const float*)d_in[6];
  const float* W1    = (const float*)d_in[7];
  const float* b1    = (const float*)d_in[8];
  const float* W2    = (const float*)d_in[9];
  const float* b2    = (const float*)d_in[10];
  const float* gamma = (const float*)d_in[11];
  const float* beta  = (const float*)d_in[12];

  constexpr int H  = 512;
  constexpr int M  = 8 * 1024;     // B*S
  constexpr int H7 = 7 * H;        // 3584
  constexpr int KG = 4 * H;        // 2048 : [h | hb | ha | c]
  constexpr int H2 = 2 * H;        // 1024

  // workspace layout (256B aligned slabs)
  char* w = (char*)d_ws;
  auto alloc = [&](size_t bytes) {
    char* p = w; w += (bytes + 255) & ~(size_t)255; return p;
  };
  unsigned short* WcatT = (unsigned short*)alloc((size_t)H7 * KG * 2); // [3584][2048]
  unsigned short* WeT   = (unsigned short*)alloc((size_t)H7 * H  * 2); // [3584][512]
  unsigned short* W1T   = (unsigned short*)alloc((size_t)H2 * H  * 2); // [1024][512]
  unsigned short* W2T   = (unsigned short*)alloc((size_t)H  * H2 * 2); // [512][1024]
  unsigned short* xb    = (unsigned short*)alloc((size_t)M * H  * 2);
  float* wec      = (float*)alloc((size_t)M * H7 * 4);                 // we_cat
  float* gatesBuf = (float*)alloc((size_t)M * H7 * 4);
  unsigned short* Abuf = (unsigned short*)alloc((size_t)M * KG * 2);
  float* cbuf_b   = (float*)alloc((size_t)M * H * 4);
  float* cbuf_a   = (float*)alloc((size_t)M * H * 4);
  float* hbuf     = (float*)alloc((size_t)M * H * 4);
  float* cbuf     = (float*)alloc((size_t)M * H * 4);
  float* lnF      = (float*)alloc((size_t)M * H * 4);
  unsigned short* lnB = (unsigned short*)alloc((size_t)M * H * 2);
  unsigned short* act = (unsigned short*)alloc((size_t)M * H2 * 2);

  // 1) weight prep (bf16, N-major)
  {
    size_t t;
    t = (size_t)(3 * H) * H7;   // Wg [1536][3584] -> WcatT rows k 0..1535
    transpose_to_bf16<<<(unsigned)((t + 255) / 256), 256, 0, stream>>>(Wg, WcatT, 3 * H, H7, KG, 0);
    t = (size_t)H * H7;         // Wp [512][3584]  -> WcatT rows k 1536..2047
    transpose_to_bf16<<<(unsigned)((t + 255) / 256), 256, 0, stream>>>(Wp, WcatT, H, H7, KG, 3 * H);
    transpose_to_bf16<<<(unsigned)((t + 255) / 256), 256, 0, stream>>>(We, WeT, H, H7, H, 0);
    t = (size_t)H * H2;
    transpose_to_bf16<<<(unsigned)((t + 255) / 256), 256, 0, stream>>>(W1, W1T, H, H2, H, 0);
    transpose_to_bf16<<<(unsigned)((t + 255) / 256), 256, 0, stream>>>(W2, W2T, H2, H, H2, 0);
  }

  // 2) init states + bf16 input
  init_states<<<(M * H + 255) / 256, 256, 0, stream>>>(word, mask, hbuf, cbuf, xb, M * H);

  // 3) we_cat = x @ We          [8192 x 3584]
  gemm_bf16_wmma<<<dim3(H7 / BN, M / BM), 256, 0, stream>>>(
      xb, WeT, wec, nullptr, nullptr, nullptr, nullptr, M, H7, H, 0);

  // 4) layers
  for (int layer = 0; layer < 4; ++layer) {
    shift_pack<<<dim3(2, M), 256, 0, stream>>>(hbuf, cbuf, mask, Abuf, cbuf_b, cbuf_a);
    // gates = [h|hb|ha|c] @ [Wg;Wp] + bg + we_cat
    gemm_bf16_wmma<<<dim3(H7 / BN, M / BM), 256, 0, stream>>>(
        Abuf, WcatT, gatesBuf, nullptr, bg, wec, nullptr, M, H7, KG, 1);
    gate_ln<<<M, 256, 0, stream>>>(gatesBuf, cbuf, cbuf_b, cbuf_a, mask, gamma, beta, lnF, lnB);
    // act = relu(ln @ W1 + b1)   [8192 x 1024] bf16
    gemm_bf16_wmma<<<dim3(H2 / BN, M / BM), 256, 0, stream>>>(
        lnB, W1T, nullptr, act, b1, nullptr, nullptr, M, H2, H, 2);
    // h = (act @ W2 + b2 + ln) * seq_mask   [8192 x 512]
    gemm_bf16_wmma<<<dim3(H / BN, M / BM), 256, 0, stream>>>(
        act, W2T, hbuf, nullptr, b2, lnF, mask, M, H, H2, 3);
  }

  // 5) output
  copy_out<<<(M * H + 255) / 256, 256, 0, stream>>>(hbuf, (float*)d_out, M * H);
}
